// ASD_RNN_20761871908973
// MI455X (gfx1250) — matrix operand order repaired and checked by reading, not compiled, not run on hardware
//
#include <hip/hip_runtime.h>

// ---------------------------------------------------------------------------
// Problem constants (from reference): B=64, S=64, F_IN=2048, H=512, P2=20
// ---------------------------------------------------------------------------
#define BB    64
#define SS    64
#define FIN   2048
#define HH    512
#define G4    2048   // 4*H
#define P2    20

typedef __attribute__((ext_vector_type(16))) __bf16        bf16x16;
typedef __attribute__((ext_vector_type(8)))  float         f32x8;
typedef __attribute__((ext_vector_type(8)))  unsigned int  u32x8;

// ---------------------------------------------------------------------------
// helpers
// ---------------------------------------------------------------------------
__device__ __forceinline__ unsigned short f2bf(float x) {
    unsigned int u = __builtin_bit_cast(unsigned int, x);
    unsigned int r = u + 0x7fffu + ((u >> 16) & 1u);   // round-to-nearest-even
    return (unsigned short)(r >> 16);
}
__device__ __forceinline__ unsigned int pk2(float a, float b) {
    return (unsigned int)f2bf(a) | ((unsigned int)f2bf(b) << 16);
}
__device__ __forceinline__ float sigmoidf(float x) {
    return 1.0f / (1.0f + __expf(-x));
}

// A-tile (16x32 MxK, bf16): lane L -> row L%16, K = (L/16)*16 + 0..15 contiguous per lane
__device__ __forceinline__ bf16x16 ldA_bf16(const unsigned short* __restrict__ A, int lda, int lane) {
    int m = lane & 15, h = lane >> 4;
    const uint4* p = (const uint4*)(A + (size_t)m * lda + h * 16);
    uint4 a = p[0], b = p[1];
    u32x8 r;
    r[0] = a.x; r[1] = a.y; r[2] = a.z; r[3] = a.w;
    r[4] = b.x; r[5] = b.y; r[6] = b.z; r[7] = b.w;
    return __builtin_bit_cast(bf16x16, r);
}
// B-tile (32x16 KxN) from N-major (transposed) weights: lane L -> col L%16,
// K = (L/16)*16 + 0..15 contiguous per lane.
__device__ __forceinline__ bf16x16 ldB_bf16(const unsigned short* __restrict__ Bt, int ldb, int lane) {
    int n = lane & 15, h = lane >> 4;
    const uint4* p = (const uint4*)(Bt + (size_t)n * ldb + h * 16);
    uint4 a = p[0], b = p[1];
    u32x8 r;
    r[0] = a.x; r[1] = a.y; r[2] = a.z; r[3] = a.w;
    r[4] = b.x; r[5] = b.y; r[6] = b.z; r[7] = b.w;
    return __builtin_bit_cast(bf16x16, r);
}
__device__ __forceinline__ f32x8 wmma_bf16(bf16x16 a, bf16x16 b, f32x8 c) {
    return __builtin_amdgcn_wmma_f32_16x16x32_bf16(false, a, false, b, (short)0, c, false, false);
}

// 2x2 tile WMMA inner loop: 4 loads -> 4 WMMAs per K-chunk (A reused across N,
// B reused across M).  A0/A1 are 16-row-apart tile bases; B0/B1 16-col-apart.
__device__ __forceinline__ void gemm_2x2(
    const unsigned short* __restrict__ A0, const unsigned short* __restrict__ A1,
    const unsigned short* __restrict__ B0, const unsigned short* __restrict__ B1,
    int lda, int ldb, int K, int lane,
    f32x8& c00, f32x8& c01, f32x8& c10, f32x8& c11) {
    for (int k0 = 0; k0 < K; k0 += 32) {
        __builtin_prefetch(B0 + k0 + 256, 0, 1);
        __builtin_prefetch(A0 + k0 + 256, 0, 1);
        bf16x16 a0 = ldA_bf16(A0 + k0, lda, lane);
        bf16x16 a1 = ldA_bf16(A1 + k0, lda, lane);
        bf16x16 b0 = ldB_bf16(B0 + k0, ldb, lane);
        bf16x16 b1 = ldB_bf16(B1 + k0, ldb, lane);
        c00 = wmma_bf16(a0, b0, c00);
        c01 = wmma_bf16(a0, b1, c01);
        c10 = wmma_bf16(a1, b0, c10);
        c11 = wmma_bf16(a1, b1, c11);
    }
}

// ---------------------------------------------------------------------------
// prep kernels
// ---------------------------------------------------------------------------
// dst[c*R + r] = bf16(src[r*C + c])   (src: [R][C] fp32, dst: [C][R] bf16)
__global__ void __launch_bounds__(256) transpose_cast_kernel(
    const float* __restrict__ src, unsigned short* __restrict__ dst, int R, int C) {
    int idx = blockIdx.x * 256 + threadIdx.x;
    if (idx >= R * C) return;
    int c = idx / R, r = idx - c * R;
    dst[idx] = f2bf(src[(size_t)r * C + c]);
}

// straight fp32 -> bf16 cast, 8 elements per thread, vectorized
__global__ void __launch_bounds__(256) cast_bf16_kernel(
    const float* __restrict__ src, unsigned short* __restrict__ dst, int n8) {
    int i = blockIdx.x * 256 + threadIdx.x;
    if (i >= n8) return;
    const float4* p = (const float4*)src + (size_t)i * 2;
    float4 x0 = p[0], x1 = p[1];
    uint4 r;
    r.x = pk2(x0.x, x0.y); r.y = pk2(x0.z, x0.w);
    r.z = pk2(x1.x, x1.y); r.w = pk2(x1.z, x1.w);
    ((uint4*)dst)[i] = r;
}

__global__ void __launch_bounds__(256) zero32_kernel(unsigned int* __restrict__ p, int n) {
    int i = blockIdx.x * 256 + threadIdx.x;
    if (i < n) p[i] = 0u;
}

// ---------------------------------------------------------------------------
// GEMM 1: fuse = relu(v_feat @ W_enc + b_enc) + cat_emb[category]  -> bf16
//   M=4096 (b*S+s), N=512, K=2048.  Block tile 64x128, 8 waves (2M x 4N),
//   each wave a 32x32 output via 2x2 WMMA tiles.
// ---------------------------------------------------------------------------
__global__ void __launch_bounds__(256) gemm_enc_kernel(
    const unsigned short* __restrict__ vfb, const unsigned short* __restrict__ Wt,
    const float* __restrict__ b_enc, const float* __restrict__ cat_emb,
    const int* __restrict__ category, unsigned short* __restrict__ fuse) {
    const int lane = threadIdx.x & 31;
    const int wave = threadIdx.x >> 5;
    const int mw = wave & 1, nw = wave >> 1;          // 2 x 4 wave grid
    const int m0 = blockIdx.x * 64 + mw * 32;
    const int n0 = blockIdx.y * 128 + nw * 32;
    f32x8 c00 = {}, c01 = {}, c10 = {}, c11 = {};
    const unsigned short* A0 = vfb + (size_t)m0 * FIN;
    const unsigned short* A1 = A0 + (size_t)16 * FIN;
    const unsigned short* B0 = Wt + (size_t)n0 * FIN;
    const unsigned short* B1 = B0 + (size_t)16 * FIN;
    gemm_2x2(A0, A1, B0, B1, FIN, FIN, FIN, lane, c00, c01, c10, c11);

    // all 64 rows of this block share one batch index (64-row alignment)
    const float* ce = cat_emb + (size_t)category[blockIdx.x] * HH;
    const int half = (lane >> 4) << 3;
    const int cA = n0 + (lane & 15), cB = cA + 16;
    const float beA = b_enc[cA], beB = b_enc[cB];
    const float ceA = ce[cA], ceB = ce[cB];
#pragma unroll
    for (int r = 0; r < 8; ++r) {
        int rA = m0 + r + half, rB = rA + 16;
        fuse[(size_t)rA * HH + cA] = f2bf(fmaxf(c00[r] + beA, 0.0f) + ceA);
        fuse[(size_t)rA * HH + cB] = f2bf(fmaxf(c01[r] + beB, 0.0f) + ceB);
        fuse[(size_t)rB * HH + cA] = f2bf(fmaxf(c10[r] + beA, 0.0f) + ceA);
        fuse[(size_t)rB * HH + cB] = f2bf(fmaxf(c11[r] + beB, 0.0f) + ceB);
    }
}

// ---------------------------------------------------------------------------
// GEMM 2: xg[s][b][g] = fuse @ Wx + (bx+bh)   M=4096, N=2048, K=512 -> fp32
// ---------------------------------------------------------------------------
__global__ void __launch_bounds__(256) gemm_xg_kernel(
    const unsigned short* __restrict__ fuse, const unsigned short* __restrict__ Wxt,
    const float* __restrict__ bx, const float* __restrict__ bh,
    float* __restrict__ xg) {
    const int lane = threadIdx.x & 31;
    const int wave = threadIdx.x >> 5;
    const int mw = wave & 1, nw = wave >> 1;
    const int m0 = blockIdx.x * 64 + mw * 32;
    const int n0 = blockIdx.y * 128 + nw * 32;
    f32x8 c00 = {}, c01 = {}, c10 = {}, c11 = {};
    const unsigned short* A0 = fuse + (size_t)m0 * HH;
    const unsigned short* A1 = A0 + (size_t)16 * HH;
    const unsigned short* B0 = Wxt + (size_t)n0 * HH;
    const unsigned short* B1 = B0 + (size_t)16 * HH;
    gemm_2x2(A0, A1, B0, B1, HH, HH, HH, lane, c00, c01, c10, c11);

    const int half = (lane >> 4) << 3;
    const int cA = n0 + (lane & 15), cB = cA + 16;
    const float bsA = bx[cA] + bh[cA];
    const float bsB = bx[cB] + bh[cB];
#pragma unroll
    for (int r = 0; r < 8; ++r) {
        int rA = m0 + r + half, rB = rA + 16;   // row = b*S + s
        int bA = rA >> 6, sA = rA & 63;
        int bB = rB >> 6, sB = rB & 63;
        size_t baseA = ((size_t)sA * BB + bA) * G4;
        size_t baseB = ((size_t)sB * BB + bB) * G4;
        xg[baseA + cA] = c00[r] + bsA;
        xg[baseA + cB] = c01[r] + bsB;
        xg[baseB + cA] = c10[r] + bsA;
        xg[baseB + cB] = c11[r] + bsB;
    }
}

// ---------------------------------------------------------------------------
// LSTM step: gates = xg[s] + h @ Wh ; fused-gate update, h = o*c (no tanh on c)
//   grid: 32 blocks (16 hidden cols each), 512 threads = 16 waves
//   wave (gate, m-tile) computes one 16x16 tile; gate fusion via LDS.
// ---------------------------------------------------------------------------
__global__ void __launch_bounds__(512) lstm_step_kernel(
    const unsigned short* __restrict__ h_in, unsigned short* __restrict__ h_out,
    const unsigned short* __restrict__ Wht, const float* __restrict__ xg_s,
    float* __restrict__ c_state, float* __restrict__ h_f32) {
    __shared__ float gl[4][64][16];
    const int lane = threadIdx.x & 31;
    const int wave = threadIdx.x >> 5;
    const int gate = wave >> 2, mt = wave & 3;
    const int j0 = blockIdx.x * 16;
    f32x8 acc = {};
    const unsigned short* Ar = h_in + (size_t)(mt * 16) * HH;
    const unsigned short* Bt = Wht + (size_t)(gate * HH + j0) * HH;
    for (int k0 = 0; k0 < HH; k0 += 32) {
        acc = wmma_bf16(ldA_bf16(Ar + k0, HH, lane), ldB_bf16(Bt + k0, HH, lane), acc);
    }
    const int half = (lane >> 4) << 3;
    const int n = lane & 15;
#pragma unroll
    for (int r = 0; r < 8; ++r) {
        int m = mt * 16 + r + half;
        gl[gate][m][n] = acc[r] + xg_s[(size_t)m * G4 + gate * HH + j0 + n];
    }
    __syncthreads();
    for (int idx = threadIdx.x; idx < 64 * 16; idx += 512) {
        int m = idx >> 4, nn = idx & 15;
        int j = j0 + nn;
        float iv = sigmoidf(gl[0][m][nn]);
        float fv = sigmoidf(gl[1][m][nn]);
        float ov = sigmoidf(gl[2][m][nn]);
        float gv = tanhf(gl[3][m][nn]);
        size_t o = (size_t)m * HH + j;
        float cv = fv * c_state[o] + iv * gv;
        c_state[o] = cv;
        float hv = ov * cv;          // NOTE: reference has no tanh here
        h_f32[o] = hv;
        h_out[o] = f2bf(hv);
    }
}

// ---------------------------------------------------------------------------
// Prototype distances: dist_feat[b, s*20+p] = log((d+1)/(d+1e-8)), d = ||v-p||^2
// ---------------------------------------------------------------------------
__global__ void __launch_bounds__(256) dist_kernel(
    const float* __restrict__ v_feat, const float* __restrict__ proto,
    float* __restrict__ dist_feat) {
    __shared__ float red[P2][256];
    const int bs = blockIdx.x;
    const int b = bs >> 6, s = bs & 63;
    const float* v = v_feat + (size_t)bs * FIN;
    float acc[P2];
#pragma unroll
    for (int p = 0; p < P2; ++p) acc[p] = 0.0f;
    for (int k = threadIdx.x; k < FIN; k += 256) {
        float x = v[k];
#pragma unroll
        for (int p = 0; p < P2; ++p) {
            float d = x - proto[(size_t)p * FIN + k];
            acc[p] = fmaf(d, d, acc[p]);
        }
    }
#pragma unroll
    for (int p = 0; p < P2; ++p) red[p][threadIdx.x] = acc[p];
    __syncthreads();
    for (int off = 128; off > 0; off >>= 1) {
        if (threadIdx.x < off) {
#pragma unroll
            for (int p = 0; p < P2; ++p)
                red[p][threadIdx.x] += red[p][threadIdx.x + off];
        }
        __syncthreads();
    }
    if (threadIdx.x < P2) {
        float d = red[threadIdx.x][0];
        dist_feat[(size_t)b * (SS * P2) + s * P2 + threadIdx.x] =
            logf((d + 1.0f) / (d + 1e-8f));
    }
}

// ---------------------------------------------------------------------------
// Final blend: out = sigmoid(h@W_dec+b)*gate + sigmoid(dist@W_dd+b)*(1-gate)
// gate = sigmoid(mean_s(v)@W_gate + b).   one block per batch element.
// ---------------------------------------------------------------------------
__global__ void __launch_bounds__(256) final_kernel(
    const float* __restrict__ h, const float* __restrict__ W_dec, const float* __restrict__ b_dec,
    const float* __restrict__ dist_feat, const float* __restrict__ W_dd, const float* __restrict__ b_dd,
    const float* __restrict__ v_feat, const float* __restrict__ W_gate, const float* __restrict__ b_gate,
    float* __restrict__ out) {
    __shared__ float r0[256], r1[256], r2[256];
    const int b = blockIdx.x, tid = threadIdx.x;
    float s0 = 0.0f, s1 = 0.0f, s2 = 0.0f;
    for (int k = tid; k < HH; k += 256) s0 += h[(size_t)b * HH + k] * W_dec[k];
    for (int i = tid; i < SS * P2; i += 256) s1 += dist_feat[(size_t)b * (SS * P2) + i] * W_dd[i];
    const float* vb = v_feat + (size_t)b * SS * FIN;
    for (int i = tid; i < SS * FIN; i += 256) s2 += vb[i] * W_gate[i & (FIN - 1)];
    r0[tid] = s0; r1[tid] = s1; r2[tid] = s2;
    __syncthreads();
    for (int off = 128; off > 0; off >>= 1) {
        if (tid < off) { r0[tid] += r0[tid + off]; r1[tid] += r1[tid + off]; r2[tid] += r2[tid + off]; }
        __syncthreads();
    }
    if (tid == 0) {
        float outp = sigmoidf(r0[0] + b_dec[0]);
        float dp   = sigmoidf(r1[0] + b_dd[0]);
        float gate = sigmoidf(r2[0] * (1.0f / (float)SS) + b_gate[0]);
        out[b] = outp * gate + dp * (1.0f - gate);
    }
}

// ---------------------------------------------------------------------------
// launcher
// ---------------------------------------------------------------------------
extern "C" void kernel_launch(void* const* d_in, const int* in_sizes, int n_in,
                              void* d_out, int out_size, void* d_ws, size_t ws_size,
                              hipStream_t stream) {
    (void)in_sizes; (void)n_in; (void)out_size; (void)ws_size;

    const float* v_feat    = (const float*)d_in[0];
    const int*   category  = (const int*)d_in[1];
    const float* W_enc     = (const float*)d_in[2];
    const float* b_enc     = (const float*)d_in[3];
    const float* Wx        = (const float*)d_in[4];
    const float* bx        = (const float*)d_in[5];
    const float* Wh        = (const float*)d_in[6];
    const float* bh        = (const float*)d_in[7];
    const float* cat_emb   = (const float*)d_in[8];
    const float* W_dec     = (const float*)d_in[9];
    const float* b_dec     = (const float*)d_in[10];
    const float* prototype = (const float*)d_in[11];
    const float* W_dd      = (const float*)d_in[12];
    const float* b_dd      = (const float*)d_in[13];
    const float* W_gate    = (const float*)d_in[14];
    const float* b_gate    = (const float*)d_in[15];
    float* out = (float*)d_out;

    // workspace carve-out (256B aligned)
    char* ws = (char*)d_ws;
    size_t off = 0;
    auto carve = [&](size_t bytes) -> void* {
        void* p = ws + off;
        off += (bytes + 255) & ~(size_t)255;
        return p;
    };
    unsigned short* vfb    = (unsigned short*)carve((size_t)BB * SS * FIN * 2); // bf16 v_feat
    unsigned short* Wenc_t = (unsigned short*)carve((size_t)HH * FIN * 2);      // [512][2048]
    unsigned short* Wx_t   = (unsigned short*)carve((size_t)G4 * HH * 2);       // [2048][512]
    unsigned short* Wh_t   = (unsigned short*)carve((size_t)G4 * HH * 2);       // [2048][512]
    unsigned short* fuse   = (unsigned short*)carve((size_t)BB * SS * HH * 2);
    float*          xg     = (float*)carve((size_t)SS * BB * G4 * 4);
    unsigned short* hbuf0  = (unsigned short*)carve((size_t)BB * HH * 2);
    unsigned short* hbuf1  = (unsigned short*)carve((size_t)BB * HH * 2);
    float*          cstate = (float*)carve((size_t)BB * HH * 4);
    float*          h_f32  = (float*)carve((size_t)BB * HH * 4);
    float*          dfeat  = (float*)carve((size_t)BB * SS * P2 * 4);

    // 1) activation cast + weight transposes to bf16 N-major
    cast_bf16_kernel<<<(BB * SS * FIN / 8 + 255) / 256, 256, 0, stream>>>(
        v_feat, vfb, BB * SS * FIN / 8);
    transpose_cast_kernel<<<(FIN * HH + 255) / 256, 256, 0, stream>>>(W_enc, Wenc_t, FIN, HH);
    transpose_cast_kernel<<<(HH * G4 + 255) / 256, 256, 0, stream>>>(Wx, Wx_t, HH, G4);
    transpose_cast_kernel<<<(HH * G4 + 255) / 256, 256, 0, stream>>>(Wh, Wh_t, HH, G4);

    // 2) zero h0 (bf16) and c0 (fp32)
    zero32_kernel<<<(BB * HH / 2 + 255) / 256, 256, 0, stream>>>((unsigned int*)hbuf0, BB * HH / 2);
    zero32_kernel<<<(BB * HH + 255) / 256, 256, 0, stream>>>((unsigned int*)cstate, BB * HH);

    // 3) encoder GEMM  (M=4096, N=512)
    gemm_enc_kernel<<<dim3(BB * SS / 64, HH / 128), 256, 0, stream>>>(
        vfb, Wenc_t, b_enc, cat_emb, category, fuse);

    // 4) input-gate GEMM (M=4096, N=2048)
    gemm_xg_kernel<<<dim3(BB * SS / 64, G4 / 128), 256, 0, stream>>>(
        fuse, Wx_t, bx, bh, xg);

    // 5) LSTM recurrence: 64 sequential steps, ping-pong h buffers
    for (int s = 0; s < SS; ++s) {
        const unsigned short* hi = (s & 1) ? hbuf1 : hbuf0;
        unsigned short*       ho = (s & 1) ? hbuf0 : hbuf1;
        lstm_step_kernel<<<HH / 16, 512, 0, stream>>>(
            hi, ho, Wh_t, xg + (size_t)s * BB * G4, cstate, h_f32);
    }

    // 6) prototype distance features
    dist_kernel<<<BB * SS, 256, 0, stream>>>(v_feat, prototype, dfeat);

    // 7) decoder + gate + blend
    final_kernel<<<BB, 256, 0, stream>>>(
        h_f32, W_dec, b_dec, dfeat, W_dd, b_dd, v_feat, W_gate, b_gate, out);
}
